// GraphNetEncoder_21784074125519
// MI455X (gfx1250) — compile-verified, matrix operand order checked by reading, not run on hardware
//
#include <hip/hip_runtime.h>

#define NN 50000
#define NE 200000

typedef __attribute__((ext_vector_type(16))) __bf16 v16bf;
typedef __attribute__((ext_vector_type(8)))  __bf16 v8bf;
typedef __attribute__((ext_vector_type(8)))  float  v8f;

enum { MODE_ENC = 0, MODE_EDGE = 1, MODE_NODE = 2, MODE_GLOB = 3 };

struct BlockArgs {
  const float*  x;        // ENC source (f32, row stride DIN)
  const __bf16* e_in;     // E x 128
  const __bf16* n_in;     // N x 128
  const __bf16* g_in;     // 128
  const int*    senders;
  const int*    receivers;
  const float*  sent;     // N x 64 f32
  const float*  recv;     // N x 64 f32
  const float*  nsum;     // 64
  const float*  esum;     // 64
  const __bf16* w0t;      // 64 x DIN (transposed, bf16)
  const float*  b0;
  const __bf16* w1t;      // 64 x 64
  const float*  b1;
  const float*  scale;
  const float*  offset;
  float*        out;      // M x 64 f32
  int           M;
};

__device__ __forceinline__ v8bf ld8bf(const __bf16* p) { return *(const v8bf*)p; }

__device__ __forceinline__ v8bf cvt8(const float* p) {
  v8bf r;
#pragma unroll
  for (int i = 0; i < 8; ++i) r[i] = (__bf16)p[i];
  return r;
}

__device__ __forceinline__ v16bf mk16(v8bf lo, v8bf hi) {
  v16bf r;
#pragma unroll
  for (int i = 0; i < 8; ++i) { r[i] = lo[i]; r[i + 8] = hi[i]; }
  return r;
}

// xor-shuffle within aligned 8-lane groups via DPP8 (pure VALU; no LDS traffic)
__host__ __device__ constexpr int dpp8sel(int m) {
  int s = 0;
  for (int i = 0; i < 8; ++i) s |= ((i ^ m) & 7) << (3 * i);
  return s;
}
template <int M>
__device__ __forceinline__ float dpp_xor(float x) {
  int xi = __builtin_bit_cast(int, x);
  int r  = __builtin_amdgcn_mov_dpp8(xi, dpp8sel(M));
  return __builtin_bit_cast(float, r);
}

// Fused: Y = GroupNorm(relu(relu(X@W0+b0)@W1+b1)), 8 groups of 8 channels.
// 128 threads = 4 waves; wave w owns output cols [16w,16w+16); WG does 64 rows.
template <int DIN, int MODE>
__global__ __launch_bounds__(128) void block_kernel(BlockArgs a) {
  const int tid  = threadIdx.x;
  const int wave = tid >> 5;
  const int lane = tid & 31;
  const int half = lane >> 4;     // 0: K {0..7,16..23}; 1: K {8..15,24..31}
  const int l16  = lane & 15;
  const int koff = half * 8;
  const int col  = wave * 16 + l16;

  __shared__ __bf16 hid[16 * 64];

  const int NRT = (MODE == MODE_GLOB) ? 1 : 4;

  const float bias0 = a.b0[col];
  const float bias1 = a.b1[col];
  const float sc    = a.scale[col];
  const float of    = a.offset[col];

  // layer-1 B fragments are row-tile invariant: load once.
  const __bf16* w1p = a.w1t + (long)col * 64 + koff;
  const v16bf B10 = mk16(ld8bf(w1p), ld8bf(w1p + 16));
  const v16bf B11 = mk16(ld8bf(w1p + 32), ld8bf(w1p + 48));

  for (int rt = 0; rt < NRT; ++rt) {
    const int rowbase = blockIdx.x * 64 + rt * 16;
    int row  = rowbase + l16;
    int rowc = row < a.M ? row : a.M - 1;   // clamp: keep EXEC all-1 for WMMA

    const float*  px = nullptr;
    const __bf16 *pe = nullptr, *ps = nullptr, *pr = nullptr, *pn = nullptr;
    const float  *pse = nullptr, *pre = nullptr;
    if constexpr (MODE == MODE_ENC)  { px = a.x + (long)rowc * DIN; }
    if constexpr (MODE == MODE_EDGE) {
      pe = a.e_in + (long)rowc * 128;
      ps = a.n_in + (long)a.senders[rowc] * 128;
      pr = a.n_in + (long)a.receivers[rowc] * 128;
    }
    if constexpr (MODE == MODE_NODE) {
      pn  = a.n_in + (long)rowc * 128;
      pse = a.sent + (long)rowc * 64;
      pre = a.recv + (long)rowc * 64;
    }

    // kc is a compile-time K base (multiple of 16 after full unroll); the
    // per-lane +koff (0 or 8) never crosses a 64/128-aligned segment boundary.
    auto loadAhalf = [&](int kc) -> v8bf {
      if constexpr (MODE == MODE_ENC) {
        return cvt8(px + kc + koff);
      } else if constexpr (MODE == MODE_EDGE) {
        const int seg = kc >> 7;
        const int kk  = (kc & 127) + koff;
        const __bf16* p = (seg == 0) ? pe : (seg == 1) ? ps : (seg == 2) ? pr : a.g_in;
        return ld8bf(p + kk);
      } else if constexpr (MODE == MODE_NODE) {
        if (kc < 128)      return ld8bf(pn + kc + koff);
        else if (kc < 192) return cvt8(pse + (kc - 128) + koff);
        else if (kc < 256) return cvt8(pre + (kc - 192) + koff);
        else               return ld8bf(a.g_in + (kc - 256) + koff);
      } else {  // MODE_GLOB: row-independent (M == 1)
        if (kc < 64)       return cvt8(a.nsum + kc + koff);
        else if (kc < 128) return cvt8(a.esum + (kc - 64) + koff);
        else               return ld8bf(a.g_in + (kc - 128) + koff);
      }
    };

    // ---- layer 0: [16 x DIN] @ [DIN x 64] (this wave: 16 cols) ----
    v8f acc = {};
#pragma unroll
    for (int ks = 0; ks < DIN / 32; ++ks) {
      const int kb = ks * 32;
      v16bf A = mk16(loadAhalf(kb), loadAhalf(kb + 16));
      const __bf16* wpt = a.w0t + (long)col * DIN + kb + koff;
      v16bf B = mk16(ld8bf(wpt), ld8bf(wpt + 16));
      acc = __builtin_amdgcn_wmma_f32_16x16x32_bf16(false, A, false, B,
                                                    (short)0, acc, false, false);
    }
#pragma unroll
    for (int r = 0; r < 8; ++r) {
      float v = acc[r] + bias0;
      v = v > 0.f ? v : 0.f;
      hid[(r + half * 8) * 64 + col] = (__bf16)v;   // D layout: M=r+8*half, N=col
    }
    __syncthreads();

    // ---- layer 1: [16 x 64] @ [64 x 64] from LDS ----
    const __bf16* hr = &hid[l16 * 64];
    v16bf A0 = mk16(ld8bf(hr + koff),      ld8bf(hr + 16 + koff));
    v16bf A1 = mk16(ld8bf(hr + 32 + koff), ld8bf(hr + 48 + koff));
    v8f acc2 = {};
    acc2 = __builtin_amdgcn_wmma_f32_16x16x32_bf16(false, A0, false, B10,
                                                   (short)0, acc2, false, false);
    acc2 = __builtin_amdgcn_wmma_f32_16x16x32_bf16(false, A1, false, B11,
                                                   (short)0, acc2, false, false);

    // ---- bias + relu + GroupNorm (groups of 8 channels == 8 lanes) ----
#pragma unroll
    for (int r = 0; r < 8; ++r) {
      float v = acc2[r] + bias1;
      v = v > 0.f ? v : 0.f;
      float s = v, q = v * v;
      s += dpp_xor<1>(s); q += dpp_xor<1>(q);
      s += dpp_xor<2>(s); q += dpp_xor<2>(q);
      s += dpp_xor<4>(s); q += dpp_xor<4>(q);
      float mu  = s * 0.125f;
      float var = q * 0.125f - mu * mu;
      float y   = (v - mu) * rsqrtf(var + 1e-5f) * sc + of;
      int orow = rowbase + r + half * 8;
      if (orow < a.M) a.out[(long)orow * 64 + col] = y;
    }
    __syncthreads();
  }
}

// W (K x 64, f32) -> Wt (64 x K, bf16)
__global__ void wprep_kernel(const float* __restrict__ w, __bf16* __restrict__ wt, int din) {
  int idx = blockIdx.x * 256 + threadIdx.x;
  if (idx < 64 * din) {
    int n = idx / din, k = idx % din;
    wt[idx] = (__bf16)w[(long)k * 64 + n];
  }
}

// dst[M x 128] bf16 = concat(x0[M x 64], x1[M x 64]) f32
__global__ void concat2_kernel(__bf16* __restrict__ dst, const float* __restrict__ x0,
                               const float* __restrict__ x1, long M) {
  long idx = (long)blockIdx.x * 256 + threadIdx.x;
  if (idx < M * 128) {
    long row = idx >> 7;
    int  c   = (int)(idx & 127);
    dst[idx] = (__bf16)(c < 64 ? x0[row * 64 + c] : x1[row * 64 + c - 64]);
  }
}

__global__ void scatter_add_kernel(const float* __restrict__ e_new, const int* __restrict__ snd,
                                   const int* __restrict__ rcv, float* sent, float* recv, int E) {
  long idx = (long)blockIdx.x * 256 + threadIdx.x;
  if (idx < (long)E * 64) {
    int e = (int)(idx >> 6), c = (int)(idx & 63);
    float v = e_new[idx];
    atomicAdd(sent + (long)snd[e] * 64 + c, v);
    atomicAdd(recv + (long)rcv[e] * 64 + c, v);
  }
}

__global__ void colsum_kernel(const float* __restrict__ x, float* out, int M) {
  int c  = threadIdx.x;           // 64 threads
  int r0 = blockIdx.x * 512;
  int r1 = r0 + 512 < M ? r0 + 512 : M;
  float s = 0.f;
  for (int r = r0; r < r1; ++r) s += x[(long)r * 64 + c];
  atomicAdd(out + c, s);
}

extern "C" void kernel_launch(void* const* d_in, const int* in_sizes, int n_in,
                              void* d_out, int out_size, void* d_ws, size_t ws_size,
                              hipStream_t stream) {
  (void)in_sizes; (void)n_in; (void)out_size; (void)ws_size;
  const float* nodes     = (const float*)d_in[0];
  const float* edges     = (const float*)d_in[1];
  const float* gfeat     = (const float*)d_in[2];
  const int*   senders   = (const int*)d_in[3];
  const int*   receivers = (const int*)d_in[4];
  float*       d_out_f   = (float*)d_out;

  // ---- carve workspace ----
  char* wsp = (char*)d_ws;
  auto alloc = [&](size_t bytes) { char* p = wsp; wsp += (bytes + 255) & ~(size_t)255; return p; };
  float*  enc_n = (float*)alloc((size_t)NN * 64 * 4);
  float*  enc_e = (float*)alloc((size_t)NE * 64 * 4);
  float*  enc_g = (float*)alloc(64 * 4);
  float*  out_n = (float*)alloc((size_t)NN * 64 * 4);
  float*  out_e = (float*)alloc((size_t)NE * 64 * 4);
  float*  out_g = (float*)alloc(64 * 4);
  float*  sent  = (float*)alloc((size_t)NN * 64 * 4);
  float*  recv  = (float*)alloc((size_t)NN * 64 * 4);
  float*  nsum  = (float*)alloc(64 * 4);
  float*  esum  = (float*)alloc(64 * 4);
  __bf16* nin_b = (__bf16*)alloc((size_t)NN * 128 * 2);
  __bf16* ein_b = (__bf16*)alloc((size_t)NE * 128 * 2);
  __bf16* gin_b = (__bf16*)alloc(128 * 2);

  // blocks: node_enc, edge_enc, global_enc, rec0{edge,node,glob}, rec1{edge,node,glob}
  struct BlkInfo { int base; int din; };
  const BlkInfo blk[9] = {{5, 32},  {11, 32}, {17, 32},
                          {23, 512}, {29, 384}, {35, 256},
                          {41, 512}, {47, 384}, {53, 256}};
  __bf16* w0t[9]; __bf16* w1t[9];
  for (int i = 0; i < 9; ++i) {
    w0t[i] = (__bf16*)alloc((size_t)64 * blk[i].din * 2);
    w1t[i] = (__bf16*)alloc((size_t)64 * 64 * 2);
  }
  for (int i = 0; i < 9; ++i) {
    int n0 = 64 * blk[i].din;
    wprep_kernel<<<(n0 + 255) / 256, 256, 0, stream>>>((const float*)d_in[blk[i].base], w0t[i], blk[i].din);
    wprep_kernel<<<(4096 + 255) / 256, 256, 0, stream>>>((const float*)d_in[blk[i].base + 2], w1t[i], 64);
  }

  auto mkargs = [&](int b, const __bf16* W0, const __bf16* W1, float* out, int M) {
    BlockArgs a{};
    a.w0t = W0; a.b0 = (const float*)d_in[b + 1];
    a.w1t = W1; a.b1 = (const float*)d_in[b + 3];
    a.scale = (const float*)d_in[b + 4]; a.offset = (const float*)d_in[b + 5];
    a.out = out; a.M = M;
    return a;
  };

  const int NB_N = (NN + 63) / 64;   // 782
  const int NB_E = (NE + 63) / 64;   // 3125

  // ---- encoders ----
  { BlockArgs a = mkargs(5, w0t[0], w1t[0], enc_n, NN); a.x = nodes;
    block_kernel<32, MODE_ENC><<<NB_N, 128, 0, stream>>>(a); }
  { BlockArgs a = mkargs(11, w0t[1], w1t[1], enc_e, NE); a.x = edges;
    block_kernel<32, MODE_ENC><<<NB_E, 128, 0, stream>>>(a); }
  { BlockArgs a = mkargs(17, w0t[2], w1t[2], enc_g, 1); a.x = gfeat;
    block_kernel<32, MODE_ENC><<<1, 128, 0, stream>>>(a); }

  // ---- recurrences ----
  const float* curn = enc_n; const float* cure = enc_e; const float* curg = enc_g;
  for (int i = 0; i < 2; ++i) {
    concat2_kernel<<<(int)(((long)NN * 128 + 255) / 256), 256, 0, stream>>>(nin_b, curn, enc_n, NN);
    concat2_kernel<<<(int)(((long)NE * 128 + 255) / 256), 256, 0, stream>>>(ein_b, cure, enc_e, NE);
    concat2_kernel<<<1, 256, 0, stream>>>(gin_b, curg, enc_g, 1);

    float* oe = (i == 1) ? d_out_f + (size_t)NN * 64 : out_e;
    float* on = (i == 1) ? d_out_f : out_n;
    float* og = (i == 1) ? d_out_f + (size_t)(NN + NE) * 64 : out_g;
    const int ei = (i == 0) ? 3 : 6, ni = (i == 0) ? 4 : 7, gi = (i == 0) ? 5 : 8;

    { BlockArgs a = mkargs(blk[ei].base, w0t[ei], w1t[ei], oe, NE);
      a.e_in = ein_b; a.n_in = nin_b; a.g_in = gin_b;
      a.senders = senders; a.receivers = receivers;
      block_kernel<512, MODE_EDGE><<<NB_E, 128, 0, stream>>>(a); }

    (void)hipMemsetAsync(sent, 0, (size_t)NN * 64 * 4, stream);
    (void)hipMemsetAsync(recv, 0, (size_t)NN * 64 * 4, stream);
    scatter_add_kernel<<<(int)(((long)NE * 64 + 255) / 256), 256, 0, stream>>>(oe, senders, receivers, sent, recv, NE);

    { BlockArgs a = mkargs(blk[ni].base, w0t[ni], w1t[ni], on, NN);
      a.n_in = nin_b; a.g_in = gin_b; a.sent = sent; a.recv = recv;
      block_kernel<384, MODE_NODE><<<NB_N, 128, 0, stream>>>(a); }

    (void)hipMemsetAsync(nsum, 0, 64 * 4, stream);
    (void)hipMemsetAsync(esum, 0, 64 * 4, stream);
    colsum_kernel<<<(NN + 511) / 512, 64, 0, stream>>>(on, nsum, NN);
    colsum_kernel<<<(NE + 511) / 512, 64, 0, stream>>>(oe, esum, NE);

    { BlockArgs a = mkargs(blk[gi].base, w0t[gi], w1t[gi], og, 1);
      a.nsum = nsum; a.esum = esum; a.g_in = gin_b;
      block_kernel<256, MODE_GLOB><<<1, 128, 0, stream>>>(a); }

    curn = on; cure = oe; curg = og;
  }
}
